// PNANet_63539746177577
// MI455X (gfx1250) — matrix operand order repaired and checked by reading, not compiled, not run on hardware
//
#include <hip/hip_runtime.h>
#include <hip/hip_bf16.h>

// ---------------------------------------------------------------------------
// PNA GNN for MI455X (gfx1250, wave32, WMMA).
//
// Heavy math (pre-MLP, post-MLP, lin) runs as v_wmma_f32_16x16x32_f16 with
// fp32 accumulation. Scalers amp/att are exactly 1.0 (constant in-degree K),
// so the 3 post-weight blocks are pre-summed on device, shrinking the post
// GEMM K-dim from 13*F' to F+4F'. Everything fuses per 16-node tile:
// gather -> preGEMM(x7 neighbors, agg in C-frags) -> postGEMM -> linGEMM.
//
// B matrices are pre-packed into WMMA register order [ktile][ntile][lane][16]
// (one 32B per-lane load per fragment). A fragments come from LDS tiles
// (ds_load_b128 pairs). Phase-1 tile loads use GLOBAL_LOAD_ASYNC_TO_LDS_B128
// when the toolchain exposes the builtin (ASYNCcnt path), else plain loads.
// ---------------------------------------------------------------------------

typedef __attribute__((ext_vector_type(16))) _Float16 v16h;
typedef __attribute__((ext_vector_type(8)))  float    v8f;
typedef __attribute__((__vector_size__(16))) int      v4i;

#define NNODES 65536
#define NGRAPH 128
#define MNODES 512
#define KNBR   7
#define NTOW   4

#if __has_builtin(__builtin_amdgcn_global_load_async_to_lds_b128)
#define HAVE_ASYNC_LDS 1
typedef __attribute__((address_space(1))) v4i* gv4i_p;  // global src (AS1)
typedef __attribute__((address_space(3))) v4i* lv4i_p;  // LDS dst (AS3)
#else
#define HAVE_ASYNC_LDS 0
#endif

__device__ __forceinline__ void wait_async0() {
#if __has_builtin(__builtin_amdgcn_s_wait_asynccnt)
  __builtin_amdgcn_s_wait_asynccnt(0);
#else
  asm volatile("s_wait_asynccnt 0" ::: "memory");
#endif
}

__device__ __forceinline__ v8f wmma16(v16h a, v16h b, v8f c) {
  // D = A(16x32 f16) * B(32x16 f16) + C(16x16 f32)
  return __builtin_amdgcn_wmma_f32_16x16x32_f16(
      /*neg_a=*/false, a, /*neg_b=*/false, b,
      /*c_mod=*/(short)0, c, /*reuse_a=*/false, /*reuse_b=*/false);
}

// A-fragment (16x32, f16) from a row-major tile; ISA 7.12.2:
// row = lane&15; lanes<16 hold K 0-7 & 16-23, lanes>=16 hold K 8-15 & 24-31.
__device__ __forceinline__ v16h load_A_one(const _Float16* p, int ld, int colBase) {
  int lane = threadIdx.x & 31;
  int row  = lane & 15;
  int kb   = (lane >> 4) << 3;  // 0 or 8
  v16h a;
#pragma unroll
  for (int i = 0; i < 16; ++i) {
    int c = colBase + kb + (i & 7) + ((i >> 3) << 4);
    a[i] = p[row * ld + c];
  }
  return a;
}

// A-fragment whose logical columns [0,W1) come from p1 (stride W1) and
// [W1,W1+W2) from p2 (stride W2); columns beyond are zero (K padding).
template <int W1, int W2>
__device__ __forceinline__ v16h load_A_pair(const _Float16* p1, const _Float16* p2,
                                            int colBase) {
  int lane = threadIdx.x & 31;
  int row  = lane & 15;
  int kb   = (lane >> 4) << 3;
  v16h a;
#pragma unroll
  for (int i = 0; i < 16; ++i) {
    int c = colBase + kb + (i & 7) + ((i >> 3) << 4);
    _Float16 v = (_Float16)0.0f;
    if (c < W1)            v = p1[row * W1 + c];
    else if (c < W1 + W2)  v = p2[row * W2 + (c - W1)];
    a[i] = v;
  }
  return a;
}

// B-fragment from pre-packed weights: one 32B contiguous load per lane.
__device__ __forceinline__ v16h load_B_packed(const _Float16* __restrict__ pkBase) {
  int lane = threadIdx.x & 31;
  return *(const v16h*)(pkBase + lane * 16);
}

// ---------------------------------------------------------------------------
// kNN: one block per graph, one thread per node; insertion-sorted top-7.
// ---------------------------------------------------------------------------
__global__ __launch_bounds__(MNODES) void knn_kernel(const float* __restrict__ pos,
                                                     int* __restrict__ nbr) {
  __shared__ float sx[MNODES], sy[MNODES], sz[MNODES];
  int g   = blockIdx.x;
  int m   = threadIdx.x;
  int gid = g * MNODES + m;
  float px = pos[gid * 3 + 0], py = pos[gid * 3 + 1], pz = pos[gid * 3 + 2];
  sx[m] = px; sy[m] = py; sz[m] = pz;
  __syncthreads();
  float bd[KNBR]; int bi[KNBR];
#pragma unroll
  for (int k = 0; k < KNBR; ++k) { bd[k] = 3.0e38f; bi[k] = 0; }
  for (int j = 0; j < MNODES; ++j) {
    float dx = px - sx[j], dy = py - sy[j], dz = pz - sz[j];
    float d2 = dx * dx + dy * dy + dz * dz;
    if (j == m) continue;  // loop=False (self excluded)
    if (d2 < bd[KNBR - 1]) {
      int p = KNBR - 1;
      while (p > 0 && bd[p - 1] > d2) { bd[p] = bd[p - 1]; bi[p] = bi[p - 1]; --p; }
      bd[p] = d2; bi[p] = j;
    }
  }
#pragma unroll
  for (int k = 0; k < KNBR; ++k) nbr[gid * KNBR + k] = g * MNODES + bi[k];
}

// ---------------------------------------------------------------------------
// Weight prep: f32 -> f16 convert; post-MLP weight fold (3 scaler blocks sum,
// since amp == att == 1.0 exactly for constant degree K), zero-padded to KAP;
// then pack into WMMA B-fragment register order.
// ---------------------------------------------------------------------------
__global__ void cvt_f16_kernel(const float* __restrict__ in, _Float16* __restrict__ out,
                               int n) {
  int i = blockIdx.x * blockDim.x + threadIdx.x;
  if (i < n) out[i] = (_Float16)in[i];
}

__global__ void combine_post_kernel(const float* __restrict__ postW,
                                    _Float16* __restrict__ wcomb,
                                    int F, int F1, int FT, int KAP) {
  int i = blockIdx.x * blockDim.x + threadIdx.x;
  int total = NTOW * KAP * FT;
  if (i >= total) return;
  int t   = i / (KAP * FT);
  int rem = i - t * KAP * FT;
  int r   = rem / FT;
  int c   = rem - r * FT;
  int R   = F + 12 * F1;  // 13*F1 rows of original post_w
  float v = 0.0f;
  if (r < F) {
    v = postW[((size_t)t * R + r) * FT + c];
  } else if (r < F + 4 * F1) {
    int rr = r - F;
    v = postW[((size_t)t * R + F + rr) * FT + c]
      + postW[((size_t)t * R + F + 4 * F1 + rr) * FT + c]   // * amp (==1)
      + postW[((size_t)t * R + F + 8 * F1 + rr) * FT + c];  // * att (==1)
  }
  wcomb[i] = (_Float16)v;
}

// Pack a row-major f16 [R,C] matrix into [R/32][C/16][lane:32][e:16] so a
// B fragment is one contiguous 32B load per lane.
// B layout (matches A): col = lane&15; lanes<16 hold K 0-7 & 16-23, lanes>=16 K 8-15 & 24-31.
__global__ void pack_B_kernel(const _Float16* __restrict__ w,
                              _Float16* __restrict__ pk, int R, int C) {
  int i = blockIdx.x * blockDim.x + threadIdx.x;
  if (i >= R * C) return;
  int idx  = i;
  int e    = idx & 15;  idx >>= 4;
  int lane = idx & 31;  idx >>= 5;
  int nt   = idx % (C / 16);
  int kt   = idx / (C / 16);
  int col  = nt * 16 + (lane & 15);
  int row  = kt * 32 + ((lane >> 4) << 3) + (e & 7) + ((e >> 3) << 4);
  pk[i] = w[(size_t)row * C + col];
}

// ---------------------------------------------------------------------------
// Fused PNAConv: block = 16-node tile (128 thr = 4 waves = 4 towers).
//   F  : input feature width      F1 : tower width (pre output)
//   FT : post output per tower    D  : NTOW*FT (layer output width)
// ---------------------------------------------------------------------------
template <int F, int F1, int FT>
__global__ __launch_bounds__(128) void pna_conv_kernel(
    const _Float16* __restrict__ xin,    // [N,F] f16
    const int* __restrict__ nbr,         // [N,K]
    const _Float16* __restrict__ preWpk, // packed [T*2F/32][F1/16][32][16]
    const float* __restrict__ preB,      // [T,F1]
    const _Float16* __restrict__ wpostpk,// packed [T*KAP/32][FT/16][32][16]
    const float* __restrict__ postB,     // [T,FT]
    const _Float16* __restrict__ linWpk, // packed [D/32][D/16][32][16]
    const float* __restrict__ linB,      // [D]
    float* __restrict__ hout)            // [N,D] f32 (pre-BN)
{
  constexpr int D   = NTOW * FT;
  constexpr int KA  = F + 4 * F1;
  constexpr int KAP = (KA + 31) & ~31;  // K padded to 32 for WMMA
  constexpr int NT1 = F1 / 16;
  constexpr int NT2 = FT / 16;
  constexpr int NT3 = D / 64;           // lin col-tiles per wave
  constexpr int KT1 = (2 * F) / 32;     // pre GEMM k-tiles
  constexpr int KT2 = KAP / 32;         // post GEMM k-tiles

  __shared__ alignas(16) _Float16 xi[16 * F];                 // center feats
  __shared__ alignas(16) _Float16 xj[KNBR * 16 * F];          // gathered nbrs
  __shared__ alignas(16) _Float16 aggLds[NTOW * 16 * 4 * F1]; // per-tower agg
  __shared__ alignas(16) _Float16 ytile[16 * D];              // post outputs

  const int lane  = threadIdx.x & 31;
  const int wave  = threadIdx.x >> 5;  // tower id
  const int t     = wave;
  const int node0 = blockIdx.x * 16;

  // Phase 1: center tile + 7 gathered neighbor tiles into LDS.
#if HAVE_ASYNC_LDS
  {
    constexpr int CH_XI = (16 * F) / 8;  // 16B chunks
    for (int c = threadIdx.x; c < CH_XI; c += 128)
      __builtin_amdgcn_global_load_async_to_lds_b128(
          (gv4i_p)(xin + (size_t)node0 * F + c * 8), (lv4i_p)(xi + c * 8), 0, 0);
    constexpr int CHR = F / 8;           // 16B chunks per gathered row
    for (int c = threadIdx.x; c < KNBR * 16 * CHR; c += 128) {
      int k   = c / (16 * CHR);
      int rem = c - k * 16 * CHR;
      int r   = rem / CHR;
      int grp = rem - r * CHR;
      int src = nbr[(node0 + r) * KNBR + k];
      __builtin_amdgcn_global_load_async_to_lds_b128(
          (gv4i_p)(xin + (size_t)src * F + grp * 8),
          (lv4i_p)(xj + (size_t)(k * 16 + r) * F + grp * 8), 0, 0);
    }
    wait_async0();
  }
#else
  for (int idx = threadIdx.x; idx < 16 * F; idx += 128)
    xi[idx] = xin[(size_t)node0 * F + idx];
  for (int idx = threadIdx.x; idx < KNBR * 16 * F; idx += 128) {
    int k   = idx / (16 * F);
    int rem = idx - k * 16 * F;
    int r   = rem / F;
    int c   = rem - r * F;
    int src = nbr[(node0 + r) * KNBR + k];
    xj[idx] = xin[(size_t)src * F + c];
  }
#endif
  __syncthreads();

  // Phase 2: per-tower pre-MLP over 7 neighbors, aggregate in C-fragments.
  // B fragments are loop-invariant across neighbors; hoist into registers.
  v16h bpre[KT1][NT1];
#pragma unroll
  for (int kt = 0; kt < KT1; ++kt)
#pragma unroll
    for (int nt = 0; nt < NT1; ++nt)
      bpre[kt][nt] = load_B_packed(preWpk + ((size_t)(t * KT1 + kt) * NT1 + nt) * 512);

  v8f vsum[NT1], vmin[NT1], vmax[NT1], vsq[NT1];
  for (int k = 0; k < KNBR; ++k) {
    v8f m[NT1] = {};
#pragma unroll
    for (int kt = 0; kt < KT1; ++kt) {
      v16h a = load_A_pair<F, F>(xi, &xj[k * 16 * F], kt * 32);
#pragma unroll
      for (int nt = 0; nt < NT1; ++nt) m[nt] = wmma16(a, bpre[kt][nt], m[nt]);
    }
#pragma unroll
    for (int nt = 0; nt < NT1; ++nt) {
      float bias = preB[t * F1 + nt * 16 + (lane & 15)];
#pragma unroll
      for (int e = 0; e < 8; ++e) m[nt][e] += bias;
      if (k == 0) {
        vsum[nt] = m[nt]; vmin[nt] = m[nt]; vmax[nt] = m[nt]; vsq[nt] = m[nt] * m[nt];
      } else {
        vsum[nt] += m[nt];
        vsq[nt]  += m[nt] * m[nt];
#pragma unroll
        for (int e = 0; e < 8; ++e) {
          vmin[nt][e] = fminf(vmin[nt][e], m[nt][e]);
          vmax[nt][e] = fmaxf(vmax[nt][e], m[nt][e]);
        }
      }
    }
  }

  // Phase 3: finalize mean/std, stage agg to LDS as f16 A-tile rows.
  _Float16* myagg = &aggLds[t * 16 * 4 * F1];
#pragma unroll
  for (int nt = 0; nt < NT1; ++nt) {
#pragma unroll
    for (int e = 0; e < 8; ++e) {
      int row = e + ((lane >> 4) << 3);  // C layout: vgpr e, lane-half -> row
      int col = nt * 16 + (lane & 15);
      float mean = vsum[nt][e] * (1.0f / KNBR);
      float ex2  = vsq[nt][e] * (1.0f / KNBR);
      float sd   = sqrtf(fmaxf(ex2 - mean * mean, 0.0f) + 1e-5f);
      myagg[row * 4 * F1 + 0 * F1 + col] = (_Float16)mean;
      myagg[row * 4 * F1 + 1 * F1 + col] = (_Float16)vmin[nt][e];
      myagg[row * 4 * F1 + 2 * F1 + col] = (_Float16)vmax[nt][e];
      myagg[row * 4 * F1 + 3 * F1 + col] = (_Float16)sd;
    }
  }
  __syncthreads();

  // Phase 4: post-MLP with folded weights, A = [xi | agg] (16 x KAP).
  v8f p[NT2] = {};
#pragma unroll
  for (int kt = 0; kt < KT2; ++kt) {
    v16h a = load_A_pair<F, 4 * F1>(xi, myagg, kt * 32);
#pragma unroll
    for (int nt = 0; nt < NT2; ++nt) {
      v16h b = load_B_packed(wpostpk + ((size_t)(t * KT2 + kt) * NT2 + nt) * 512);
      p[nt] = wmma16(a, b, p[nt]);
    }
  }
#pragma unroll
  for (int nt = 0; nt < NT2; ++nt) {
    float bias = postB[t * FT + nt * 16 + (lane & 15)];
#pragma unroll
    for (int e = 0; e < 8; ++e) {
      int row = e + ((lane >> 4) << 3);
      int col = t * FT + nt * 16 + (lane & 15);
      ytile[row * D + col] = (_Float16)(p[nt][e] + bias);
    }
  }
  __syncthreads();

  // Phase 5: lin GEMM [16,D] x [D,D]; D/16 col-tiles split across 4 waves.
  v8f o[NT3] = {};
#pragma unroll
  for (int kt = 0; kt < D / 32; ++kt) {
    v16h a = load_A_one(ytile, D, kt * 32);
#pragma unroll
    for (int nt = 0; nt < NT3; ++nt) {
      int ct = wave * NT3 + nt;
      v16h b = load_B_packed(linWpk + ((size_t)kt * (D / 16) + ct) * 512);
      o[nt] = wmma16(a, b, o[nt]);
    }
  }
#pragma unroll
  for (int nt = 0; nt < NT3; ++nt) {
    int cb = (wave * NT3 + nt) * 16 + (lane & 15);
    float bias = linB[cb];
#pragma unroll
    for (int e = 0; e < 8; ++e) {
      int row = e + ((lane >> 4) << 3);
      hout[(size_t)(node0 + row) * D + cb] = o[nt][e] + bias;
    }
  }
}

// ---------------------------------------------------------------------------
// BatchNorm (training stats), deterministic two-pass reduction (no atomics).
// ---------------------------------------------------------------------------
__global__ __launch_bounds__(256) void bn_stats_partial(const float* __restrict__ h,
                                                        float* __restrict__ ps,
                                                        float* __restrict__ pq,
                                                        int n, int D) {
  int j    = threadIdx.x % D;
  int sub  = threadIdx.x / D;
  int nsub = blockDim.x / D;
  int rpb  = n / gridDim.x;
  int r0   = blockIdx.x * rpb;
  float s = 0.0f, q = 0.0f;
  for (int r = r0 + sub; r < r0 + rpb; r += nsub) {
    float v = h[(size_t)r * D + j];
    s += v; q += v * v;
  }
  __shared__ float ls[256], lq[256];
  ls[threadIdx.x] = s; lq[threadIdx.x] = q;
  __syncthreads();
  if (sub == 0) {
    for (int u = 1; u < nsub; ++u) { s += ls[u * D + j]; q += lq[u * D + j]; }
    ps[blockIdx.x * D + j] = s;
    pq[blockIdx.x * D + j] = q;
  }
}

__global__ void bn_reduce(const float* __restrict__ ps, const float* __restrict__ pq,
                          float* __restrict__ sums, float* __restrict__ sumsq,
                          int D, int nblk) {
  int j = blockIdx.x * blockDim.x + threadIdx.x;
  if (j >= D) return;
  float s = 0.0f, q = 0.0f;
  for (int b = 0; b < nblk; ++b) { s += ps[b * D + j]; q += pq[b * D + j]; }
  sums[j] = s; sumsq[j] = q;
}

__global__ void bn_apply(const float* __restrict__ h, const float* __restrict__ sums,
                         const float* __restrict__ sumsq, const float* __restrict__ gam,
                         const float* __restrict__ bet, _Float16* __restrict__ out,
                         int n, int D) {
  int i = blockIdx.x * blockDim.x + threadIdx.x;
  if (i >= n * D) return;
  int j = i % D;
  float inv_n = 1.0f / (float)n;
  float mu  = sums[j] * inv_n;
  float var = sumsq[j] * inv_n - mu * mu;
  float v   = gam[j] * (h[i] - mu) * rsqrtf(var + 1e-5f) + bet[j];
  out[i] = (_Float16)fmaxf(v, 0.0f);
}

// Global mean pool: one block per graph, thread = feature column.
__global__ __launch_bounds__(128) void pool_kernel(const _Float16* __restrict__ x,
                                                   float* __restrict__ out) {
  int g = blockIdx.x, j = threadIdx.x;
  float s = 0.0f;
  for (int r = 0; r < MNODES; ++r) s += (float)x[(size_t)(g * MNODES + r) * 128 + j];
  out[g * 128 + j] = s * (1.0f / MNODES);
}

// ---------------------------------------------------------------------------
extern "C" void kernel_launch(void* const* d_in, const int* in_sizes, int n_in,
                              void* d_out, int out_size, void* d_ws, size_t ws_size,
                              hipStream_t stream) {
  (void)in_sizes; (void)n_in; (void)out_size; (void)ws_size;
  const float* x       = (const float*)d_in[0];
  const float* pos     = (const float*)d_in[1];
  // d_in[2] = batch (contiguous per graph; layout known, unused)
  const float* pre1_w  = (const float*)d_in[3];
  const float* pre1_b  = (const float*)d_in[4];
  const float* post1_w = (const float*)d_in[5];
  const float* post1_b = (const float*)d_in[6];
  const float* lin1_w  = (const float*)d_in[7];
  const float* lin1_b  = (const float*)d_in[8];
  const float* bn1_g   = (const float*)d_in[9];
  const float* bn1_b   = (const float*)d_in[10];
  const float* pre2_w  = (const float*)d_in[11];
  const float* pre2_b  = (const float*)d_in[12];
  const float* post2_w = (const float*)d_in[13];
  const float* post2_b = (const float*)d_in[14];
  const float* lin2_w  = (const float*)d_in[15];
  const float* lin2_b  = (const float*)d_in[16];
  const float* bn2_g   = (const float*)d_in[17];
  const float* bn2_b   = (const float*)d_in[18];

  char* base = (char*)d_ws;
  size_t off = 0;
  auto carve = [&](size_t bytes) -> void* {
    void* p = base + off;
    off += (bytes + 255) & ~(size_t)255;
    return p;
  };

  int*       nbr      = (int*)      carve((size_t)NNODES * KNBR * 4);
  _Float16*  x16_1    = (_Float16*) carve((size_t)NNODES * 16 * 2);
  _Float16*  x16_2    = (_Float16*) carve((size_t)NNODES * 64 * 2);
  _Float16*  pool_in  = (_Float16*) carve((size_t)NNODES * 128 * 2);
  float*     h1       = (float*)    carve((size_t)NNODES * 64 * 4);
  float*     h2       = (float*)    carve((size_t)NNODES * 128 * 4);
  _Float16*  preW16_1 = (_Float16*) carve((size_t)NTOW * 32 * 16 * 2);
  _Float16*  preW16_2 = (_Float16*) carve((size_t)NTOW * 128 * 64 * 2);
  _Float16*  wcomb1   = (_Float16*) carve((size_t)NTOW * 96 * 16 * 2);   // KAP=96
  _Float16*  wcomb2   = (_Float16*) carve((size_t)NTOW * 320 * 32 * 2);  // KAP=320
  _Float16*  lin16_1  = (_Float16*) carve((size_t)64 * 64 * 2);
  _Float16*  lin16_2  = (_Float16*) carve((size_t)128 * 128 * 2);
  _Float16*  prePk1   = (_Float16*) carve((size_t)NTOW * 32 * 16 * 2);
  _Float16*  prePk2   = (_Float16*) carve((size_t)NTOW * 128 * 64 * 2);
  _Float16*  postPk1  = (_Float16*) carve((size_t)NTOW * 96 * 16 * 2);
  _Float16*  postPk2  = (_Float16*) carve((size_t)NTOW * 320 * 32 * 2);
  _Float16*  linPk1   = (_Float16*) carve((size_t)64 * 64 * 2);
  _Float16*  linPk2   = (_Float16*) carve((size_t)128 * 128 * 2);
  float*     ps       = (float*)    carve((size_t)128 * 128 * 4);
  float*     pq       = (float*)    carve((size_t)128 * 128 * 4);
  float*     sums     = (float*)    carve((size_t)128 * 4);
  float*     sumsq    = (float*)    carve((size_t)128 * 4);

  auto cdiv = [](int a, int b) { return (a + b - 1) / b; };

  // Graph build + weight/activation prep (f16 + fragment packing).
  knn_kernel<<<NGRAPH, MNODES, 0, stream>>>(pos, nbr);
  cvt_f16_kernel<<<cdiv(NNODES * 16, 256), 256, 0, stream>>>(x, x16_1, NNODES * 16);
  cvt_f16_kernel<<<cdiv(NTOW * 32 * 16, 256), 256, 0, stream>>>(pre1_w, preW16_1, NTOW * 32 * 16);
  cvt_f16_kernel<<<cdiv(64 * 64, 256), 256, 0, stream>>>(lin1_w, lin16_1, 64 * 64);
  cvt_f16_kernel<<<cdiv(NTOW * 128 * 64, 256), 256, 0, stream>>>(pre2_w, preW16_2, NTOW * 128 * 64);
  cvt_f16_kernel<<<cdiv(128 * 128, 256), 256, 0, stream>>>(lin2_w, lin16_2, 128 * 128);
  combine_post_kernel<<<cdiv(NTOW * 96 * 16, 256), 256, 0, stream>>>(post1_w, wcomb1, 16, 16, 16, 96);
  combine_post_kernel<<<cdiv(NTOW * 320 * 32, 256), 256, 0, stream>>>(post2_w, wcomb2, 64, 64, 32, 320);

  pack_B_kernel<<<cdiv(NTOW * 32 * 16, 256), 256, 0, stream>>>(preW16_1, prePk1, NTOW * 32, 16);
  pack_B_kernel<<<cdiv(NTOW * 128 * 64, 256), 256, 0, stream>>>(preW16_2, prePk2, NTOW * 128, 64);
  pack_B_kernel<<<cdiv(NTOW * 96 * 16, 256), 256, 0, stream>>>(wcomb1, postPk1, NTOW * 96, 16);
  pack_B_kernel<<<cdiv(NTOW * 320 * 32, 256), 256, 0, stream>>>(wcomb2, postPk2, NTOW * 320, 32);
  pack_B_kernel<<<cdiv(64 * 64, 256), 256, 0, stream>>>(lin16_1, linPk1, 64, 64);
  pack_B_kernel<<<cdiv(128 * 128, 256), 256, 0, stream>>>(lin16_2, linPk2, 128, 128);

  // Layer 1: PNAConv(F=16 -> 64) + BN + ReLU.
  pna_conv_kernel<16, 16, 16><<<NNODES / 16, 128, 0, stream>>>(
      x16_1, nbr, prePk1, pre1_b, postPk1, post1_b, linPk1, lin1_b, h1);
  bn_stats_partial<<<128, 256, 0, stream>>>(h1, ps, pq, NNODES, 64);
  bn_reduce<<<1, 64, 0, stream>>>(ps, pq, sums, sumsq, 64, 128);
  bn_apply<<<cdiv(NNODES * 64, 256), 256, 0, stream>>>(h1, sums, sumsq, bn1_g, bn1_b,
                                                       x16_2, NNODES, 64);

  // Layer 2: PNAConv(F=64 -> 128) + BN + ReLU.
  pna_conv_kernel<64, 64, 32><<<NNODES / 16, 128, 0, stream>>>(
      x16_2, nbr, prePk2, pre2_b, postPk2, post2_b, linPk2, lin2_b, h2);
  bn_stats_partial<<<128, 256, 0, stream>>>(h2, ps, pq, NNODES, 128);
  bn_reduce<<<1, 128, 0, stream>>>(ps, pq, sums, sumsq, 128, 128);
  bn_apply<<<cdiv(NNODES * 128, 256), 256, 0, stream>>>(h2, sums, sumsq, bn2_g, bn2_b,
                                                        pool_in, NNODES, 128);

  // Global mean pool -> [B,128] f32.
  pool_kernel<<<NGRAPH, 128, 0, stream>>>(pool_in, (float*)d_out);
}
